// SynthesisLayer_75780402971124
// MI455X (gfx1250) — matrix operand order repaired and checked by reading, not compiled, not run on hardware
//
#include <hip/hip_runtime.h>
#include <hip/hip_bf16.h>

// ---------------- problem constants ----------------
#define BB    8
#define CINC  512
#define COUTC 512
#define WDIMC 512
#define SINC  64
#define OHC   66            // conv output spatial (64 + 2*2 - 2)
#define TAPS  12
#define SLOPE 0.2f
#define GAINF 1.41421356237309515f
#define CLAMPF 256.0f

// LDS activation tile: 3 rows x 82 cols, each cell = 32 bf16 ci = 16 dwords,
// padded to 20 dwords (80 B) so the 16-lane B-fragment reads are bank-conflict-free.
#define CELL    20
#define TILE_P  (3 * 82 * CELL)       // padded dwords per buffer (19680 B)
#define TILE_L  (3 * 82 * 16)         // logical dwords of real data

// ---------------- workspace layout (bytes, all 256-aligned) ----------------
#define OFF_STYLES 0u
#define OFF_SSUM   16384u
#define OFF_S      16640u
#define OFF_DEMOD  33024u
#define OFF_Q      49408u                 // 512*512 f32 = 1 MB
#define OFF_WT     1097984u               // 9*512*512 bf16 = 4.5 MB
#define OFF_XP     5816576u               // 8*64*64*512 bf16 (NHWC) = 32 MB
#define OFF_Y      39371008u              // 8*512*66*66 f32 = 68 MB

// ---------------- WMMA vector types (gfx1250) ----------------
typedef __attribute__((ext_vector_type(16))) __bf16 v16bf;
typedef __attribute__((ext_vector_type(8)))  float  v8f;

union FragU { int4 i2[2]; v16bf v; };

// ---------------- async global->LDS availability (probe via __has_builtin) ----------------
#if defined(__HIP_DEVICE_COMPILE__) && defined(__has_builtin)
#if __has_builtin(__builtin_amdgcn_global_load_async_to_lds_b128) && \
    __has_builtin(__builtin_amdgcn_s_wait_asynccnt)
#define USE_ASYNC_LDS 1
#endif
#endif
#ifndef USE_ASYNC_LDS
#define USE_ASYNC_LDS 0
#endif

#if USE_ASYNC_LDS
typedef int v4i __attribute__((vector_size(16)));        // native clang vector, matches builtin
typedef __attribute__((address_space(1))) v4i gas_i4;    // global v4i32
typedef __attribute__((address_space(3))) v4i lds_i4;    // LDS v4i32
#endif

__device__ __forceinline__ unsigned short f2bf(float f) {
    unsigned int u = __float_as_uint(f);
    u += 0x7FFFu + ((u >> 16) & 1u);      // round-to-nearest-even
    return (unsigned short)(u >> 16);
}

// ============ styles[b][i] = (w[b] . affine_w[i]) / sqrt(512) + affine_b[i] ============
__global__ __launch_bounds__(256)
void k_styles(const float* __restrict__ w, const float* __restrict__ aw,
              const float* __restrict__ ab, float* __restrict__ styles) {
    int gid = blockIdx.x * 256 + threadIdx.x;        // 4096
    int b = gid >> 9, i = gid & 511;
    const float* wr = w + b * WDIMC;
    const float* ar = aw + i * WDIMC;
    float acc = 0.f;
    for (int d = 0; d < WDIMC; ++d) acc += wr[d] * ar[d];
    styles[gid] = acc * 0.04419417382415922f + ab[i];
}

// ============ sum(styles^2) over all 4096 ============
__global__ __launch_bounds__(256)
void k_ssum(const float* __restrict__ styles, float* __restrict__ ssum) {
    __shared__ float red[256];
    int tid = threadIdx.x;
    float p = 0.f;
    for (int j = tid; j < BB * CINC; j += 256) { float v = styles[j]; p += v * v; }
    red[tid] = p; __syncthreads();
    for (int s = 128; s > 0; s >>= 1) { if (tid < s) red[tid] += red[tid + s]; __syncthreads(); }
    if (tid == 0) ssum[0] = red[0];
}

// ============ s[b][i] = styles * rsqrt(mean(styles^2)) ============
__global__ __launch_bounds__(256)
void k_snorm(const float* __restrict__ styles, const float* __restrict__ ssum,
             float* __restrict__ s) {
    int gid = blockIdx.x * 256 + threadIdx.x;
    float r = rsqrtf(ssum[0] * (1.0f / 4096.0f));
    s[gid] = styles[gid] * r;
}

// ============ per-o weight norm: wt_bf16[kh][kw][o][i], q[o][i]=rn^2*sum_k cw^2 ============
__global__ __launch_bounds__(256)
void k_wnorm(const float* __restrict__ cw, float* __restrict__ q,
             unsigned short* __restrict__ wt) {
    __shared__ float red[256];
    int o = blockIdx.x, tid = threadIdx.x;
    const float* co = cw + o * (CINC * 9);
    float p = 0.f;
    for (int j = tid; j < CINC * 9; j += 256) { float v = co[j]; p += v * v; }
    red[tid] = p; __syncthreads();
    for (int s = 128; s > 0; s >>= 1) { if (tid < s) red[tid] += red[tid + s]; __syncthreads(); }
    float rn = rsqrtf(red[0] * (1.0f / 4608.0f));
    for (int i = tid; i < CINC; i += 256) {
        float qs = 0.f;
        #pragma unroll
        for (int k2 = 0; k2 < 9; ++k2) {
            float v = co[i * 9 + k2];
            qs += v * v;
            wt[(k2 * COUTC + o) * CINC + i] = f2bf(v * rn);
        }
        q[o * CINC + i] = qs * rn * rn;
    }
}

// ============ demod[b][o] = rsqrt(sum_i s^2 * q[o][i] + 1e-8) ============
__global__ __launch_bounds__(256)
void k_demod(const float* __restrict__ s, const float* __restrict__ q,
             float* __restrict__ demod) {
    int gid = blockIdx.x * 256 + threadIdx.x;        // 4096
    int b = gid & 7, o = gid >> 3;
    const float* sr = s + b * CINC;
    const float* qr = q + o * CINC;
    float acc = 1e-8f;
    for (int i = 0; i < CINC; ++i) { float si = sr[i]; acc += si * si * qr[i]; }
    demod[b * COUTC + o] = rsqrtf(acc);
}

// ============ pack x' = bf16(x * s) in NHWC [b][iy][ix][ci] ============
__global__ __launch_bounds__(256)
void k_pack_x(const float* __restrict__ x, const float* __restrict__ s,
              unsigned int* __restrict__ xp) {
    __shared__ unsigned short tile[64][34];
    int iy = blockIdx.x, b = blockIdx.y, tid = threadIdx.x;
    int ixr = tid & 63, c0 = tid >> 6;       // read role
    int ixw = tid >> 2, dq = tid & 3;        // write role
    for (int cc = 0; cc < CINC; cc += 32) {
        #pragma unroll
        for (int k = 0; k < 8; ++k) {
            int ci = cc + c0 * 8 + k;
            float v = x[((b * CINC + ci) * SINC + iy) * SINC + ixr] * s[b * CINC + ci];
            tile[ixr][c0 * 8 + k] = f2bf(v);
        }
        __syncthreads();
        const unsigned int* srcT = (const unsigned int*)&tile[ixw][0];
        unsigned int* dst = xp + ((b * SINC + iy) * SINC + ixw) * 256 + (cc >> 1) + dq * 4;
        #pragma unroll
        for (int m = 0; m < 4; ++m) dst[m] = srcT[dq * 4 + m];
        __syncthreads();
    }
}

// -------- staging helpers (fallback reg path) --------
__device__ __forceinline__ void load_chunk_regs(const unsigned int* __restrict__ xp,
                                                int b, int y, int ci0, int tid,
                                                unsigned int* val) {
    #pragma unroll
    for (int j = 0; j < 16; ++j) {
        int d = tid + j * 256;
        unsigned int v = 0u;
        if (d < TILE_L) {
            int r   = d / (82 * 16);
            int rem = d - r * (82 * 16);
            int col = rem >> 4;
            int dp  = rem & 15;
            int iy = y - 2 + r;
            int ix = col - 2;
            if (iy >= 0 && iy < SINC && ix >= 0 && ix < SINC)
                v = xp[((b * SINC + iy) * SINC + ix) * 256 + (ci0 >> 1) + dp];
        }
        val[j] = v;
    }
}

__device__ __forceinline__ void store_chunk_lds(unsigned int* __restrict__ buf, int tid,
                                                const unsigned int* val) {
    #pragma unroll
    for (int j = 0; j < 16; ++j) {
        int d = tid + j * 256;
        if (d < TILE_L) buf[(d >> 4) * CELL + (d & 15)] = val[j];   // padded cell layout
    }
}

#if USE_ASYNC_LDS
// -------- async staging: thread (col2=tid>>2, q=tid&3) copies one b128 per valid row --------
__device__ __forceinline__ void issue_async_chunk(const unsigned int* __restrict__ xp,
                                                  unsigned int* __restrict__ buf,
                                                  int b, int y, int ci0, int tid) {
    const int col2 = tid >> 2, q = tid & 3;
    #pragma unroll
    for (int r = 0; r < 3; ++r) {
        int iy = y - 2 + r;
        if (iy >= 0 && iy < SINC) {
            unsigned long long goff =
                (unsigned long long)(((b * SINC + iy) * SINC + col2)) * 512ull +
                (unsigned long long)(ci0 * 2 + q * 16);
            gas_i4* gp = (gas_i4*)((unsigned long long)(const char*)xp + goff);
            lds_i4* lp = (lds_i4*)((char*)buf + ((r * 82 + col2 + 2) * CELL + q * 4) * 4);
            __builtin_amdgcn_global_load_async_to_lds_b128(gp, lp, 0, 0);
        }
    }
}
#endif

// ============ WMMA implicit-GEMM 3x3 conv (double-buffered LDS) ============
// block: 256 thr = 8 waves; tile M=128 couts x N=80 pixels (one output row y)
// Y[b][o][y][x] = demod[b][o] * sum_{ci,kh,kw} wt[o,ci,kh,kw]*x'[b,ci,y+kh-2,x+kw-2] + cb[o]
__global__ __launch_bounds__(256)
void k_conv_wmma(const unsigned int* __restrict__ xp,      // NHWC bf16 as dwords
                 const unsigned short* __restrict__ wt,    // [kk][o][ci] bf16
                 const float* __restrict__ demod,
                 const float* __restrict__ cb,
                 float* __restrict__ Y) {
    __shared__ __align__(16) unsigned int XsD[2 * TILE_P];   // double buffer, 2 x 19.7 KB
    const int y  = blockIdx.x;     // 0..65
    const int ot = blockIdx.y;     // 0..3
    const int b  = blockIdx.z;
    const int tid  = threadIdx.x;
    const int wv   = tid >> 5;
    const int lane = tid & 31;
    const int hl   = lane >> 4;    // which 16-lane half
    const int l16  = lane & 15;
    const int o0   = ot * 128 + wv * 16;

    v8f acc[5] = {};

    // per-wave weight base (K rows o0+l16, ci columns)
    const unsigned short* wbase = wt + (o0 + l16) * CINC + hl * 8;

#if USE_ASYNC_LDS
    // zero both buffers once: padding columns/dps and out-of-range rows stay zero forever
    for (int d = tid; d < 2 * TILE_P; d += 256) XsD[d] = 0u;
    __syncthreads();
    issue_async_chunk(xp, XsD, b, y, 0, tid);
#else
    unsigned int val[16];
    load_chunk_regs(xp, b, y, 0, tid, val);
    store_chunk_lds(XsD, tid, val);
    __syncthreads();
#endif

    for (int c = 0; c < CINC / 32; ++c) {
        const int ci0 = c * 32;
        const unsigned int* Xb = XsD + (c & 1) * TILE_P;

#if USE_ASYNC_LDS
        __builtin_amdgcn_s_wait_asynccnt(0);     // chunk c resident in LDS (this wave's copies)
        __syncthreads();                         // ... and everyone else's
        if (c + 1 < CINC / 32) {
            issue_async_chunk(xp, XsD + ((c + 1) & 1) * TILE_P, b, y, ci0 + 32, tid);
            __builtin_prefetch(wbase + ci0 + 32, 0, 3);
        }
#else
        if (c + 1 < CINC / 32) {
            load_chunk_regs(xp, b, y, ci0 + 32, tid, val);
            __builtin_prefetch(wbase + ci0 + 32, 0, 3);
        }
#endif

        #pragma unroll
        for (int kk = 0; kk < 9; ++kk) {
            const int kh = kk / 3, kw = kk - kh * 3;
            // A fragment: 16 couts x 32 ci (ISA 16-bit A layout: lane<16 K0-7,16-23; lane>=16 K8-15,24-31)
            FragU fa;
            const unsigned short* wp = wbase + kk * (COUTC * CINC) + ci0;
            fa.i2[0] = *(const int4*)(wp);
            fa.i2[1] = *(const int4*)(wp + 16);
            #pragma unroll
            for (int nt = 0; nt < 5; ++nt) {
                // B fragment: 32 ci x 16 pixels (lane<16 K0-7,16-23; lane>=16 K8-15,24-31)
                int col = nt * 16 + l16 + kw;                 // 0..81
                FragU fb;
                const unsigned int* bp = Xb + (kh * 82 + col) * CELL + hl * 4;
                fb.i2[0] = *(const int4*)(bp);
                fb.i2[1] = *(const int4*)(bp + 8);
                acc[nt] = __builtin_amdgcn_wmma_f32_16x16x32_bf16(
                    false, fa.v, false, fb.v, (short)0, acc[nt], false, false);
            }
        }

#if !USE_ASYNC_LDS
        if (c + 1 < CINC / 32)
            store_chunk_lds(XsD + ((c + 1) & 1) * TILE_P, tid, val);
        __syncthreads();
#endif
    }

    // epilogue: D layout lane l, vgpr v -> (m = v + 8*(l>>4), n = l&15)
    #pragma unroll
    for (int v = 0; v < 8; ++v) {
        int om = o0 + v + 8 * hl;
        float dm = demod[b * COUTC + om];
        float bias = cb[om];
        #pragma unroll
        for (int nt = 0; nt < 5; ++nt) {
            int xc = nt * 16 + l16;
            if (xc < OHC)
                Y[((b * COUTC + om) * OHC + y) * OHC + xc] = acc[nt][v] * dm + bias;
        }
    }
}

// ============ fused up2 -> lrelu*sqrt2 clamp -> down2 (per b,c,oy block) ============
__global__ __launch_bounds__(128)
void k_resample(const float* __restrict__ Y, const float* __restrict__ upf,
                const float* __restrict__ dnf, float* __restrict__ out) {
    __shared__ float sY[11 * 66];   // conv rows oy-4..oy+6
    __shared__ float sV[66];        // one up-H row
    __shared__ float sRH[138];      // down-H row
    __shared__ float sFu[12], sFd[12];
    const int oy = blockIdx.x, c = blockIdx.y, b = blockIdx.z;
    const int tid = threadIdx.x;

    if (tid < 12) { sFu[tid] = upf[11 - tid] * 2.0f; sFd[tid] = dnf[11 - tid]; }
    for (int idx = tid; idx < 11 * 66; idx += 128) {
        int i = idx / 66, xx = idx - i * 66;
        int gr = oy - 4 + i;
        sY[idx] = (gr >= 0 && gr < OHC) ? Y[((b * COUTC + c) * OHC + gr) * OHC + xx] : 0.f;
    }
    __syncthreads();

    float rh0 = 0.f, rh1 = 0.f;
    for (int r12 = 0; r12 < 12; ++r12) {
        int r = 2 * oy + r12;           // intermediate row in [0,138)
        if (tid < 66) {                 // up-H: V[r][x]
            float v = 0.f;
            #pragma unroll
            for (int t = 0; t < 12; ++t) {
                int j = r - 9 + t;
                if (!(j & 1)) {
                    int jh = j >> 1;
                    if (jh >= 0 && jh < OHC) v += sFu[t] * sY[(jh - oy + 4) * 66 + tid];
                }
            }
            sV[tid] = v;
        }
        __syncthreads();
        float fdr = sFd[r12];
        #pragma unroll
        for (int rep = 0; rep < 2; ++rep) {
            int xx = tid + rep * 128;
            if (xx < 138) {             // up-W + lrelu + clamp, accumulate down-H
                float u = 0.f;
                #pragma unroll
                for (int t = 0; t < 12; ++t) {
                    int jj = xx - 9 + t;
                    if (!(jj & 1)) {
                        int jh = jj >> 1;
                        if (jh >= 0 && jh < OHC) u += sFu[t] * sV[jh];
                    }
                }
                float tv = (u >= 0.f ? u : SLOPE * u) * GAINF;
                tv = fminf(fmaxf(tv, -CLAMPF), CLAMPF);
                if (rep == 0) rh0 += fdr * tv; else rh1 += fdr * tv;
            }
        }
        __syncthreads();
    }
    sRH[tid] = rh0;
    if (tid < 10) sRH[tid + 128] = rh1;
    __syncthreads();
    if (tid < 64) {                     // down-W
        float o = 0.f;
        #pragma unroll
        for (int t = 0; t < 12; ++t) o += sFd[t] * sRH[2 * tid + t];
        __builtin_nontemporal_store(o, &out[((b * COUTC + c) * 64 + oy) * 64 + tid]);
    }
}

// ================= host launcher =================
extern "C" void kernel_launch(void* const* d_in, const int* in_sizes, int n_in,
                              void* d_out, int out_size, void* d_ws, size_t ws_size,
                              hipStream_t stream) {
    const float* x   = (const float*)d_in[0];
    const float* w   = (const float*)d_in[1];
    const float* aw  = (const float*)d_in[2];
    const float* ab  = (const float*)d_in[3];
    const float* cw  = (const float*)d_in[4];
    const float* cb  = (const float*)d_in[5];
    const float* upf = (const float*)d_in[6];
    const float* dnf = (const float*)d_in[7];
    float* out = (float*)d_out;
    char* ws = (char*)d_ws;

    float*          styles = (float*)(ws + OFF_STYLES);
    float*          ssum   = (float*)(ws + OFF_SSUM);
    float*          s      = (float*)(ws + OFF_S);
    float*          demod  = (float*)(ws + OFF_DEMOD);
    float*          q      = (float*)(ws + OFF_Q);
    unsigned short* wt     = (unsigned short*)(ws + OFF_WT);
    unsigned int*   xp     = (unsigned int*)(ws + OFF_XP);
    float*          Y      = (float*)(ws + OFF_Y);

    k_styles<<<16, 256, 0, stream>>>(w, aw, ab, styles);
    k_ssum<<<1, 256, 0, stream>>>(styles, ssum);
    k_snorm<<<16, 256, 0, stream>>>(styles, ssum, s);
    k_wnorm<<<COUTC, 256, 0, stream>>>(cw, q, wt);
    k_demod<<<16, 256, 0, stream>>>(s, q, demod);
    k_pack_x<<<dim3(SINC, BB), 256, 0, stream>>>(x, s, xp);
    k_conv_wmma<<<dim3(OHC, 4, BB), 256, 0, stream>>>(xp, wt, demod, cb, Y);
    k_resample<<<dim3(64, COUTC, BB), 128, 0, stream>>>(Y, upf, dnf, out);
}
